// HypergraphTransformerLayer_7258494730647
// MI455X (gfx1250) — matrix operand order repaired and checked by reading, not compiled, not run on hardware
//
#include <hip/hip_runtime.h>

// ---------------------------------------------------------------------------
// Hypergraph transformer layer for MI455X (gfx1250, wave32, WMMA bf16).
// All matmuls use v_wmma_f32_16x16x32_bf16 (f32 accumulate). Attention is a
// split flash-attention: the bipartite mask sparsity (node rows only see edge
// keys + diagonal, edge rows only see node keys + diagonal) restricts the key
// loop, and each query tile's key range is split into 2048-key chunks so all
// 1024 wave-tasks do near-equal work (64 or 65 x 32-key tiles). Partial (O, m, l)
// results are merged by a bandwidth-trivial combine kernel.
// ---------------------------------------------------------------------------

#define NT    10240   // total tokens (nodes + edges)
#define NNODE 8192
#define NEDGE 2048
#define DH    256     // model dim
#define DFF   1024    // ffn hidden dim
#define KCHUNK 2048   // keys per flash-attention task
#define NQT   (NT / 16)        // 640 query tiles
#define NTASK (NNODE / 16 * (NEDGE / KCHUNK) + NEDGE / 16 * (NNODE / KCHUNK)) // 1024

typedef __bf16 bf16_t;
typedef __attribute__((ext_vector_type(16))) __bf16 v16bf;
typedef __attribute__((ext_vector_type(8)))  __bf16 v8bf;
typedef __attribute__((ext_vector_type(8)))  float  v8f;

static __device__ inline v8f wmma_bf16(v16bf a, v16bf b, v8f c) {
  return __builtin_amdgcn_wmma_f32_16x16x32_bf16(
      /*neg_a=*/false, a, /*neg_b=*/false, b,
      /*c_mod=*/(short)0, c, /*reuse_a=*/false, /*reuse_b=*/false);
}

static __device__ inline float redmax16(float x) {
#pragma unroll
  for (int off = 1; off < 16; off <<= 1) x = fmaxf(x, __shfl_xor(x, off, 32));
  return x;
}
static __device__ inline float redsum16(float x) {
#pragma unroll
  for (int off = 1; off < 16; off <<= 1) x += __shfl_xor(x, off, 32);
  return x;
}
static __device__ inline float redsum32(float x) {
#pragma unroll
  for (int off = 1; off < 32; off <<= 1) x += __shfl_xor(x, off, 32);
  return x;
}

// A-fragment (16x32 bf16, row-major source): lane holds row rr, K chunks at
// base=(half*8) and base+16 -> two 16B loads.
static __device__ inline v16bf load_a_frag(const bf16_t* rowbase, int rr, int half,
                                           int ld, int kbase) {
  const bf16_t* p = rowbase + (size_t)rr * ld + kbase + half * 8;
  v16bf a;
  *(v8bf*)&a = *(const v8bf*)p;
  *(((v8bf*)&a) + 1) = *(const v8bf*)(p + 16);
  return a;
}

// B-fragment (32x16 bf16) from a TRANSPOSED (Ncols x K) row-major buffer:
// lane holds column rr, K chunks at base=(half*8) and base+16 -> two 16B loads.
static __device__ inline v16bf load_b_frag(const bf16_t* colbase, int rr, int half,
                                           int ld, int kbase) {
  const bf16_t* p = colbase + (size_t)rr * ld + kbase + half * 8;
  v16bf b;
  *(v8bf*)&b = *(const v8bf*)p;
  *(((v8bf*)&b) + 1) = *(const v8bf*)(p + 16);
  return b;
}

// ---------------------------------------------------------------------------
// elementwise converts
// ---------------------------------------------------------------------------
__global__ void to_bf16_kernel(const float* __restrict__ in, bf16_t* __restrict__ out, int n) {
  int i = blockIdx.x * blockDim.x + threadIdx.x;
  if (i < n) out[i] = (bf16_t)in[i];
}

// in: (K x Nc) f32 -> out: (Nc x K) bf16 (transposed for B-fragment loads)
__global__ void transpose_bf16_kernel(const float* __restrict__ in, bf16_t* __restrict__ out,
                                      int K, int Nc) {
  int i = blockIdx.x * blockDim.x + threadIdx.x;
  if (i < K * Nc) {
    int k = i / Nc, c = i % Nc;
    out[(size_t)c * K + k] = (bf16_t)in[i];
  }
}

// ---------------------------------------------------------------------------
// Generic bf16 WMMA GEMM: C(MxNc) = A(MxK) @ B(KxNc) + bias
// A row-major bf16, Bt = B transposed (Nc x K) bf16.
// Wave computes 16x64; block = 8 waves -> 128x64 tile. M%128==0, Nc%64==0.
// ---------------------------------------------------------------------------
__global__ void gemm_wmma_kernel(const bf16_t* __restrict__ A, const bf16_t* __restrict__ Bt,
                                 const float* __restrict__ bias,
                                 float* __restrict__ Cf, bf16_t* __restrict__ Ch,
                                 int M, int Nc, int K, int relu, int storeT) {
  const int lane = threadIdx.x & 31;
  const int wave = threadIdx.x >> 5;
  const int rr = lane & 15;
  const int half = lane >> 4;
  const int row0 = blockIdx.y * 128 + wave * 16;
  const int col0 = blockIdx.x * 64;
  if (row0 >= M) return;

  v8f acc[4];
#pragma unroll
  for (int f = 0; f < 4; ++f)
#pragma unroll
    for (int e = 0; e < 8; ++e) acc[f][e] = 0.0f;

  const bf16_t* Arow = A + (size_t)row0 * K;
  for (int k = 0; k < K; k += 32) {
    v16bf a = load_a_frag(Arow, rr, half, K, k);
#pragma unroll
    for (int f = 0; f < 4; ++f) {
      v16bf b = load_b_frag(Bt + (size_t)(col0 + f * 16) * K, rr, half, K, k);
      acc[f] = wmma_bf16(a, b, acc[f]);
    }
  }

#pragma unroll
  for (int f = 0; f < 4; ++f) {
    int c = col0 + f * 16 + rr;
    float bv = bias ? bias[c] : 0.0f;
#pragma unroll
    for (int v = 0; v < 8; ++v) {
      int r = row0 + v + half * 8;
      float val = acc[f][v] + bv;
      if (relu) val = fmaxf(val, 0.0f);
      if (Cf) Cf[(size_t)r * Nc + c] = val;
      if (Ch) {
        if (storeT) Ch[(size_t)c * M + r] = (bf16_t)val;   // transposed store (for V^T)
        else        Ch[(size_t)r * Nc + c] = (bf16_t)val;
      }
    }
  }
}

// ---------------------------------------------------------------------------
// Split flash attention with bipartite + diagonal mask.
// 1024 wave-tasks, each = (16 query rows) x (2048-key chunk [+ self block]).
// Q,K row-major bf16 (NT x DH), Vt = V transposed (DH x NT) bf16.
// Emits unnormalized partial O (16x256 f32) plus per-row m, l.
// ---------------------------------------------------------------------------
static __device__ inline bool mask_ok(int i, int j, const int* __restrict__ Hm) {
  if (i == j) return true;
  if (i < NNODE) return (j >= NNODE) && (Hm[(size_t)i * NEDGE + (j - NNODE)] != 0);
  return (j < NNODE) && (Hm[(size_t)j * NEDGE + (i - NNODE)] != 0);
}

__global__ void flash_attn_split_kernel(const bf16_t* __restrict__ Q,
                                        const bf16_t* __restrict__ Kh,
                                        const bf16_t* __restrict__ Vt,
                                        const int* __restrict__ Hm,
                                        float* __restrict__ Opart,
                                        float* __restrict__ Mpart,
                                        float* __restrict__ Lpart) {
  __shared__ bf16_t ldsP[4][16 * 32];

  const int lane = threadIdx.x & 31;
  const int wave = threadIdx.x >> 5;
  const int rr = lane & 15;
  const int half = lane >> 4;
  const int task = blockIdx.x * 4 + wave;
  const float kscale = 0.0625f;  // 1/sqrt(256)
  const float NEG_INF = -__builtin_inff();

  // task -> (query tile, key chunk). Node tiles: 1 chunk (edge range == 2048).
  // Edge tiles: 4 chunks over the node range. All tasks: 64 x 32-key tiles, +1 self tile on chunk 0.
  int qt, chunk, lo;
  if (task < NNODE / 16) {          // 512 node tasks
    qt = task; chunk = 0; lo = NNODE;
  } else {                          // 512 edge tasks
    int tt = task - NNODE / 16;
    qt = NNODE / 16 + (tt >> 2); chunk = tt & 3; lo = 0;
  }
  const int i0 = qt * 16;
  const int jlo = lo + chunk * KCHUNK;
  const int jhi = jlo + KCHUNK;

  int iv[8];
#pragma unroll
  for (int v = 0; v < 8; ++v) iv[v] = i0 + v + half * 8;

  float mrow[8], lrow[8];
#pragma unroll
  for (int v = 0; v < 8; ++v) { mrow[v] = NEG_INF; lrow[v] = 0.0f; }
  v8f o[16];
#pragma unroll
  for (int f = 0; f < 16; ++f)
#pragma unroll
    for (int e = 0; e < 8; ++e) o[f][e] = 0.0f;

  bf16_t* myP = &ldsP[wave][0];
  const bf16_t* Qrow = Q + (size_t)i0 * DH;

  auto process_tile = [&](int j0) {  // 32 keys: j0 .. j0+31
    // ---- S = Q @ K^T over full D, two 16-key subtiles ----
    v8f s0, s1;
#pragma unroll
    for (int e = 0; e < 8; ++e) { s0[e] = 0.0f; s1[e] = 0.0f; }
#pragma unroll
    for (int dc = 0; dc < 8; ++dc) {
      v16bf a  = load_a_frag(Qrow, rr, half, DH, dc * 32);
      v16bf b0 = load_b_frag(Kh + (size_t)j0 * DH,        rr, half, DH, dc * 32);
      v16bf b1 = load_b_frag(Kh + (size_t)(j0 + 16) * DH, rr, half, DH, dc * 32);
      s0 = wmma_bf16(a, b0, s0);
      s1 = wmma_bf16(a, b1, s1);
    }

    // ---- mask + online softmax (row = v + 8*half, col = rr (+16)) ----
    const int ja = j0 + rr;
    const int jb = ja + 16;
    float fac[8];
#pragma unroll
    for (int v = 0; v < 8; ++v) {
      float a0 = mask_ok(iv[v], ja, Hm) ? s0[v] * kscale : NEG_INF;
      float a1 = mask_ok(iv[v], jb, Hm) ? s1[v] * kscale : NEG_INF;
      float tmax = redmax16(fmaxf(a0, a1));
      float mnew = fmaxf(mrow[v], tmax);
      // NaN-safe: fully masked chunks keep m=-inf, l=0, O=0.
      float f  = (mrow[v] == NEG_INF) ? 0.0f : __expf(mrow[v] - mnew);
      float p0 = (a0 == NEG_INF)      ? 0.0f : __expf(a0 - mnew);
      float p1 = (a1 == NEG_INF)      ? 0.0f : __expf(a1 - mnew);
      lrow[v] = lrow[v] * f + redsum16(p0 + p1);
      mrow[v] = mnew;
      fac[v] = f;
      int r = v + 8 * half;
      myP[r * 32 + rr]      = (bf16_t)p0;
      myP[r * 32 + 16 + rr] = (bf16_t)p1;
    }

    // rescale running O
#pragma unroll
    for (int f = 0; f < 16; ++f)
#pragma unroll
      for (int v = 0; v < 8; ++v) o[f][v] *= fac[v];

    // make LDS writes visible to our own ds reads (per-wave slab, in-order DS)
    asm volatile("s_wait_dscnt 0" ::: "memory");

    // reload P as A-fragment (16x32)
    v16bf pf;
    {
      const bf16_t* pp = myP + rr * 32 + half * 8;
      *(v8bf*)&pf = *(const v8bf*)pp;
      *(((v8bf*)&pf) + 1) = *(const v8bf*)(pp + 16);
    }

    // ---- O += P @ V_tile, 16 column groups of 16 ----
#pragma unroll
    for (int f = 0; f < 16; ++f) {
      v16bf b = load_b_frag(Vt + (size_t)(f * 16) * NT, rr, half, NT, j0);
      o[f] = wmma_bf16(pf, b, o[f]);
    }
  };

  // self 32-block first (chunk 0 only; guarantees finite max via diagonal)
  if (chunk == 0) process_tile(i0 & ~31);
  for (int j0 = jlo; j0 < jhi; j0 += 32) process_tile(j0);

  // ---- store unnormalized partials ----
  float* Ot = Opart + (size_t)task * (16 * DH);
#pragma unroll
  for (int f = 0; f < 16; ++f) {
    int c = f * 16 + rr;
#pragma unroll
    for (int v = 0; v < 8; ++v)
      Ot[(v + 8 * half) * DH + c] = o[f][v];
  }
  if (rr == 0) {
#pragma unroll
    for (int v = 0; v < 8; ++v) {
      Mpart[task * 16 + v + 8 * half] = mrow[v];
      Lpart[task * 16 + v + 8 * half] = lrow[v];
    }
  }
}

// ---------------------------------------------------------------------------
// Combine partial chunks: out = (sum_c w_c O_c) / (sum_c w_c l_c),
// w_c = exp(m_c - m*). One wave per output row.
// ---------------------------------------------------------------------------
__global__ void attn_combine_kernel(const float* __restrict__ Opart,
                                    const float* __restrict__ Mpart,
                                    const float* __restrict__ Lpart,
                                    bf16_t* __restrict__ Ah) {
  const int lane = threadIdx.x & 31;
  const int wave = threadIdx.x >> 5;
  const int row = blockIdx.x * 8 + wave;
  const int qt = row >> 4;
  const int r = row & 15;
  const float NEG_INF = -__builtin_inff();

  int t0, nc;
  if (qt < NNODE / 16) { t0 = qt; nc = 1; }
  else                 { t0 = NNODE / 16 + (qt - NNODE / 16) * 4; nc = 4; }

  float mstar = NEG_INF;
  for (int c = 0; c < nc; ++c) mstar = fmaxf(mstar, Mpart[(t0 + c) * 16 + r]);

  float w[4];
  float lsum = 0.0f;
  for (int c = 0; c < nc; ++c) {
    float mc = Mpart[(t0 + c) * 16 + r];
    w[c] = (mc == NEG_INF) ? 0.0f : __expf(mc - mstar);
    lsum += w[c] * Lpart[(t0 + c) * 16 + r];
  }
  float inv = 1.0f / lsum;

#pragma unroll
  for (int k = 0; k < 8; ++k) {
    int d = k * 32 + lane;
    float acc = 0.0f;
    for (int c = 0; c < nc; ++c)
      acc += w[c] * Opart[(size_t)(t0 + c) * (16 * DH) + r * DH + d];
    Ah[(size_t)row * DH + d] = (bf16_t)(acc * inv);
  }
}

// ---------------------------------------------------------------------------
// fused residual-add + LayerNorm. One wave per row (256 cols -> 8 per lane).
// ---------------------------------------------------------------------------
__global__ void add_ln_kernel(const float* __restrict__ X, const float* __restrict__ Yd,
                              const float* __restrict__ g, const float* __restrict__ b,
                              float* __restrict__ outF, bf16_t* __restrict__ outH) {
  const int lane = threadIdx.x & 31;
  const int wave = threadIdx.x >> 5;
  const int row = blockIdx.x * 8 + wave;
  const float* xr = X + (size_t)row * DH;
  const float* yr = Yd + (size_t)row * DH;

  float x[8];
  float s = 0.0f;
#pragma unroll
  for (int k = 0; k < 8; ++k) {
    int d = k * 32 + lane;
    x[k] = xr[d] + yr[d];
    s += x[k];
  }
  float mu = redsum32(s) * (1.0f / DH);
  float vs = 0.0f;
#pragma unroll
  for (int k = 0; k < 8; ++k) { float t = x[k] - mu; vs += t * t; }
  float var = redsum32(vs) * (1.0f / DH);
  float rstd = rsqrtf(var + 1e-5f);
#pragma unroll
  for (int k = 0; k < 8; ++k) {
    int d = k * 32 + lane;
    float y = (x[k] - mu) * rstd * g[d] + b[d];
    outF[(size_t)row * DH + d] = y;
    if (outH) outH[(size_t)row * DH + d] = (bf16_t)y;
  }
}

// ---------------------------------------------------------------------------
extern "C" void kernel_launch(void* const* d_in, const int* in_sizes, int n_in,
                              void* d_out, int out_size, void* d_ws, size_t ws_size,
                              hipStream_t stream) {
  const float* Z  = (const float*)d_in[0];
  const int*   Hm = (const int*)d_in[1];
  const float* Wq = (const float*)d_in[2];
  const float* bq = (const float*)d_in[3];
  const float* Wk = (const float*)d_in[4];
  const float* bk = (const float*)d_in[5];
  const float* Wv = (const float*)d_in[6];
  const float* bv = (const float*)d_in[7];
  const float* Wo = (const float*)d_in[8];
  const float* bo = (const float*)d_in[9];
  const float* W1 = (const float*)d_in[10];
  const float* b1 = (const float*)d_in[11];
  const float* W2 = (const float*)d_in[12];
  const float* b2 = (const float*)d_in[13];
  const float* g1 = (const float*)d_in[14];
  const float* be1= (const float*)d_in[15];
  const float* g2 = (const float*)d_in[16];
  const float* be2= (const float*)d_in[17];
  float* out = (float*)d_out;

  char* ws = (char*)d_ws;
  size_t off = 0;
  auto alloc = [&](size_t bytes) -> void* {
    void* p = ws + off;
    off = (off + bytes + 255) & ~(size_t)255;
    return p;
  };

  bf16_t* Zh   = (bf16_t*)alloc((size_t)NT * DH * 2);
  bf16_t* Qh   = (bf16_t*)alloc((size_t)NT * DH * 2);
  bf16_t* Kh   = (bf16_t*)alloc((size_t)NT * DH * 2);
  bf16_t* Vt   = (bf16_t*)alloc((size_t)DH * NT * 2);   // V transposed (DH x NT)
  bf16_t* Ah   = (bf16_t*)alloc((size_t)NT * DH * 2);   // attention output, bf16
  float*  AoF  = (float*)alloc((size_t)NT * DH * 4);    // after Wo, f32
  float*  Z1F  = (float*)alloc((size_t)NT * DH * 4);
  bf16_t* Z1h  = (bf16_t*)alloc((size_t)NT * DH * 2);
  bf16_t* HidH = (bf16_t*)alloc((size_t)NT * DFF * 2);
  float*  FfnF = (float*)alloc((size_t)NT * DH * 4);
  float*  Opart= (float*)alloc((size_t)NTASK * 16 * DH * 4);  // 16.8 MB
  float*  Mpart= (float*)alloc((size_t)NTASK * 16 * 4);
  float*  Lpart= (float*)alloc((size_t)NTASK * 16 * 4);
  bf16_t* Wqt  = (bf16_t*)alloc((size_t)DH * DH * 2);
  bf16_t* Wkt  = (bf16_t*)alloc((size_t)DH * DH * 2);
  bf16_t* Wvt  = (bf16_t*)alloc((size_t)DH * DH * 2);
  bf16_t* Wot  = (bf16_t*)alloc((size_t)DH * DH * 2);
  bf16_t* W1t  = (bf16_t*)alloc((size_t)DFF * DH * 2);  // (1024 x 256)
  bf16_t* W2t  = (bf16_t*)alloc((size_t)DH * DFF * 2);  // (256 x 1024)

  // ---- convert / transpose operands to bf16 ----
  {
    int n = NT * DH;
    to_bf16_kernel<<<(n + 255) / 256, 256, 0, stream>>>(Z, Zh, n);
  }
  transpose_bf16_kernel<<<(DH * DH + 255) / 256, 256, 0, stream>>>(Wq, Wqt, DH, DH);
  transpose_bf16_kernel<<<(DH * DH + 255) / 256, 256, 0, stream>>>(Wk, Wkt, DH, DH);
  transpose_bf16_kernel<<<(DH * DH + 255) / 256, 256, 0, stream>>>(Wv, Wvt, DH, DH);
  transpose_bf16_kernel<<<(DH * DH + 255) / 256, 256, 0, stream>>>(Wo, Wot, DH, DH);
  transpose_bf16_kernel<<<(DH * DFF + 255) / 256, 256, 0, stream>>>(W1, W1t, DH, DFF);
  transpose_bf16_kernel<<<(DFF * DH + 255) / 256, 256, 0, stream>>>(W2, W2t, DFF, DH);

  // ---- QKV projections ----
  dim3 g256(DH / 64, NT / 128);   // (4, 80)
  gemm_wmma_kernel<<<g256, 256, 0, stream>>>(Zh, Wqt, bq, nullptr, Qh, NT, DH, DH, 0, 0);
  gemm_wmma_kernel<<<g256, 256, 0, stream>>>(Zh, Wkt, bk, nullptr, Kh, NT, DH, DH, 0, 0);
  gemm_wmma_kernel<<<g256, 256, 0, stream>>>(Zh, Wvt, bv, nullptr, Vt, NT, DH, DH, 0, 1);

  // ---- masked flash attention (split) + combine ----
  flash_attn_split_kernel<<<NTASK / 4, 128, 0, stream>>>(Qh, Kh, Vt, Hm, Opart, Mpart, Lpart);
  attn_combine_kernel<<<NT / 8, 256, 0, stream>>>(Opart, Mpart, Lpart, Ah);

  // ---- output projection ----
  gemm_wmma_kernel<<<g256, 256, 0, stream>>>(Ah, Wot, bo, AoF, nullptr, NT, DH, DH, 0, 0);

  // ---- Z1 = LN(Z + attn_out) ----
  add_ln_kernel<<<NT / 8, 256, 0, stream>>>(Z, AoF, g1, be1, Z1F, Z1h);

  // ---- FFN ----
  dim3 g1024(DFF / 64, NT / 128);  // (16, 80)
  gemm_wmma_kernel<<<g1024, 256, 0, stream>>>(Z1h, W1t, b1, nullptr, HidH, NT, DFF, DH, 1, 0);
  gemm_wmma_kernel<<<g256, 256, 0, stream>>>(HidH, W2t, b2, FfnF, nullptr, NT, DH, DFF, 0, 0);

  // ---- out = LN(Z1 + ffn_out) ----
  add_ln_kernel<<<NT / 8, 256, 0, stream>>>(Z1F, FfnF, g2, be2, out, nullptr);
}
